// SelfAttention_40596030882361
// MI455X (gfx1250) — matrix operand order repaired and checked by reading, not compiled, not run on hardware
//
#include <hip/hip_runtime.h>

// ---------------------------------------------------------------------------
// Self-attention for MI455X (gfx1250, wave32, WMMA).
//   B=4, C=512, N=4096, CQK=64.  ~90 GFLOP, matrix-core bound -> all GEMMs on
//   v_wmma_f32_16x16x32_f16.  320KB LDS/WGP holds a full 16x4096 score
//   row-block (f16) PLUS a double-buffered async-staged V tile, so softmax is
//   fully on-chip and the big P*V GEMM streams V through LDS via
//   global_load_async_to_lds_b128 (ASYNCcnt ping-pong).
// ---------------------------------------------------------------------------

typedef _Float16 f16;
typedef _Float16 v8h  __attribute__((ext_vector_type(8)));
typedef _Float16 v16h __attribute__((ext_vector_type(16)));
typedef float    v8f  __attribute__((ext_vector_type(8)));
typedef int      v4i  __attribute__((vector_size(16)));   // matches builtin

typedef __attribute__((address_space(1))) v4i* as1_v4i;   // global b128 ptr
typedef __attribute__((address_space(3))) v4i* as3_v4i;   // LDS b128 ptr

#define B_   4
#define C_   512
#define N_   4096
#define CQK_ 64
#define SLD  (N_ + 8)   // padded LDS score row stride (f16): 8208B, kills
                        // the 8KB-stride bank aliasing, keeps 16B alignment
#define KT_  64         // k(j)-tile of the P*V GEMM staged in LDS
#define VLD_ (KT_ + 8)  // padded V-tile row stride: 144B = 36 banks,
                        // gcd(36,64)=4 -> 16 lanes hit 16 disjoint bank quads

#if __has_builtin(__builtin_amdgcn_global_load_async_to_lds_b128)
#define HAVE_ASYNC_LDS 1
#endif

static __device__ __forceinline__ v16h join8(v8h lo, v8h hi) {
  return __builtin_shufflevector(lo, hi, 0,1,2,3,4,5,6,7,8,9,10,11,12,13,14,15);
}

// A fragment, 16x32 f16 (M x K).  lane: row m = m0+(l&15);
// K = k0 + 16h + {0..15} contiguous, h=l>>4.
static __device__ __forceinline__ v16h load_a16(const f16* M, int ld, int m0,
                                                int k0, int lane) {
  const f16* p = M + (size_t)(m0 + (lane & 15)) * ld + k0 + ((lane >> 4) * 16);
  v8h lo = *(const v8h*)(p);
  v8h hi = *(const v8h*)(p + 8);
  return join8(lo, hi);
}

// B fragment, 32x16 f16 (K x N), element B[k][n] = M[n*ld + k].
// lane: col n = n0+(l&15); K = k0 + 16h + {0..15} contiguous.
static __device__ __forceinline__ v16h load_b16(const f16* M, int ld, int n0,
                                                int k0, int lane) {
  const f16* p = M + (size_t)(n0 + (lane & 15)) * ld + k0 + ((lane >> 4) * 16);
  v8h lo = *(const v8h*)(p);
  v8h hi = *(const v8h*)(p + 8);
  return join8(lo, hi);
}

static __device__ __forceinline__ v8f wmma16(v16h a, v16h b, v8f c) {
  return __builtin_amdgcn_wmma_f32_16x16x32_f16(false, a, false, b,
                                                (short)0, c, false, false);
}

// --------------------- async global -> LDS (CDNA5 path) --------------------
static __device__ __forceinline__ void cp_async_b128(f16* lds_dst,
                                                     const f16* gsrc) {
#if defined(HAVE_ASYNC_LDS)
  __builtin_amdgcn_global_load_async_to_lds_b128(
      (as1_v4i)(uintptr_t)gsrc,
      (as3_v4i)(uint32_t)(uintptr_t)lds_dst, 0, 0);
#else
  *(v8h*)lds_dst = *(const v8h*)gsrc;     // synchronous fallback
#endif
}

static __device__ __forceinline__ void wait_async0() {
#if __has_builtin(__builtin_amdgcn_s_wait_asynccnt)
  __builtin_amdgcn_s_wait_asynccnt(0);
#else
  asm volatile("s_wait_asynccnt 0x0" ::: "memory");
#endif
}

// Cooperative fill of one V tile [C_][KT_] (rows padded to VLD_).
// Thread t copies rows 2t and 2t+1, KT_/8 b128 segments each.
static __device__ __forceinline__ void fill_vtile(f16* dst, const f16* vsrc,
                                                  int r0, int kbase) {
#pragma unroll
  for (int r = 0; r < 2; ++r) {
    int c = r0 + r;
    f16* ld = dst + (size_t)c * VLD_;
    const f16* gs = vsrc + (size_t)c * N_ + kbase;
#pragma unroll
    for (int s = 0; s < KT_ / 8; ++s)
      cp_async_b128(ld + s * 8, gs + s * 8);
  }
}

// --------------------------- elementwise convert ---------------------------
__global__ void cvt_f32_f16(const float* __restrict__ src,
                            f16* __restrict__ dst, int n) {
  int i = blockIdx.x * blockDim.x + threadIdx.x;
  if (i < n) dst[i] = (f16)src[i];
}

// -------------------- x [B][C][N] f32 -> xT [B][N][C] f16 ------------------
__global__ void transpose_cvt(const float* __restrict__ x,
                              f16* __restrict__ xT) {
  __shared__ float tile[32][33];
  int b = blockIdx.z;
  int n0 = blockIdx.x * 32, c0 = blockIdx.y * 32;
  int tx = threadIdx.x, ty = threadIdx.y;          // 32 x 8
  const float* xb = x + (size_t)b * C_ * N_;
#pragma unroll
  for (int s = 0; s < 32; s += 8)
    tile[ty + s][tx] = xb[(size_t)(c0 + ty + s) * N_ + n0 + tx];
  __syncthreads();
  f16* xtb = xT + (size_t)b * N_ * C_;
#pragma unroll
  for (int s = 0; s < 32; s += 8)
    xtb[(size_t)(n0 + ty + s) * C_ + c0 + tx] = (f16)tile[tx][ty + s];
}

// ------------------------------ projections --------------------------------
// out[n][o] (mode 0, q/k position-major) or out[o][n] (mode 1, v chan-major).
// Workgroup: 8 waves, 128 rows (m) x 64 cols (o).  K = C_ = 512.
__global__ void __launch_bounds__(256)
proj_gemm(const f16* __restrict__ xT, const f16* __restrict__ W,
          const float* __restrict__ bias, f16* __restrict__ out,
          int O, int mode) {
  int b    = blockIdx.z;
  int m0   = blockIdx.x * 128 + (threadIdx.x >> 5) * 16;
  int o0   = blockIdx.y * 64;
  int lane = threadIdx.x & 31;
  const f16* A = xT + (size_t)b * N_ * C_;

  v8f acc[4] = {};
  for (int k0 = 0; k0 < C_; k0 += 32) {
    v16h a = load_a16(A, C_, m0, k0, lane);
#pragma unroll
    for (int t = 0; t < 4; ++t) {
      v16h bm = load_b16(W, C_, o0 + 16 * t, k0, lane);
      acc[t] = wmma16(a, bm, acc[t]);
    }
  }

  int h = lane >> 4, nl = lane & 15;
#pragma unroll
  for (int t = 0; t < 4; ++t) {
    int   o  = o0 + 16 * t + nl;
    float bs = bias[o];
    if (mode == 0) {                       // out[b][m][o], rows stride O
      f16* p = out + ((size_t)b * N_ + (m0 + 8 * h)) * O + o;
#pragma unroll
      for (int v = 0; v < 8; ++v) p[(size_t)v * O] = (f16)(acc[t][v] + bs);
    } else {                               // out[b][o][m], 8 contiguous f16
      v8h r;
#pragma unroll
      for (int v = 0; v < 8; ++v) r[v] = (f16)(acc[t][v] + bs);
      *(v8h*)(out + ((size_t)b * O + o) * N_ + m0 + 8 * h) = r;
    }
  }
}

// --------------------------- fused attention -------------------------------
// Grid (N/16, B), 256 threads.  LDS: S[16][SLD] f16 (~128KB) + stats +
// double-buffered V tile 2 x [512][VLD_] f16 (~144KB).  Total ~272KB of the
// 320KB WGP LDS.
__global__ void __launch_bounds__(256)
attention_fused(const f16* __restrict__ q,   // [B][N][64]
                const f16* __restrict__ k,   // [B][N][64]
                const f16* __restrict__ v,   // [B][C][N]
                float* __restrict__ out) {   // [B][C][N]
  extern __shared__ char smem_raw[];
  f16*   S     = (f16*)smem_raw;
  float* stats = (float*)(smem_raw + (size_t)16 * SLD * sizeof(f16));
  f16*   vbuf  = (f16*)(smem_raw + (size_t)16 * SLD * sizeof(f16)
                                 + 16 * sizeof(float));

  int b    = blockIdx.y;
  int i0   = blockIdx.x * 16;
  int wave = threadIdx.x >> 5;
  int lane = threadIdx.x & 31;
  int h    = lane >> 4, nl = lane & 15;

  const f16* qb = q + ((size_t)b * N_ + i0) * CQK_;
  const f16* kb = k + (size_t)b * N_ * CQK_;
  const f16* vb = v + (size_t)b * C_ * N_;

  // Q fragments (rows i0..i0+15, K=64 -> 2 frags), reused for all j tiles.
  v16h a0 = load_a16(qb, CQK_, 0, 0, lane);
  v16h a1 = load_a16(qb, CQK_, 0, 32, lane);

  // Phase 1: S[0:16][j] = Q K^T, 16-wide j tiles round-robin over waves.
  for (int jt = wave; jt < N_ / 16; jt += 8) {
    int j0 = jt * 16;
    if (jt + 8 < N_ / 16)
      __builtin_prefetch(kb + (size_t)(j0 + 128) * CQK_, 0, 2);
    v8f acc = {};
    acc = wmma16(a0, load_b16(kb, CQK_, j0, 0, lane), acc);
    acc = wmma16(a1, load_b16(kb, CQK_, j0, 32, lane), acc);
    f16* sp = S + j0 + nl;
#pragma unroll
    for (int vv = 0; vv < 8; ++vv)
      sp[(size_t)(vv + 8 * h) * SLD] = (f16)acc[vv];
  }
  __syncthreads();

  // Phase 2: per-row softmax stats; store unnormalized exp back in place.
  for (int r = wave * 2; r < wave * 2 + 2; ++r) {
    f16* row = S + (size_t)r * SLD;
    float m = -3.0e38f;
    for (int t = lane; t < N_; t += 32) m = fmaxf(m, (float)row[t]);
#pragma unroll
    for (int off = 16; off >= 1; off >>= 1)
      m = fmaxf(m, __shfl_xor(m, off, 32));
    float s = 0.f;
    for (int t = lane; t < N_; t += 32) {
      float e = __expf((float)row[t] - m);
      s += e;
      row[t] = (f16)e;
    }
#pragma unroll
    for (int off = 16; off >= 1; off >>= 1) s += __shfl_xor(s, off, 32);
    if (lane == 0) stats[r] = 1.0f / s;
  }
  __syncthreads();

  // Phase 3: O[c][i] = sum_j P[i][j] v[c][j].  K-dim = 4096, tiled by KT_,
  // V tiles double-buffered in LDS via async copies (ASYNCcnt ping-pong).
  int c0 = wave * 64;                      // each wave: 64 output channels
  int r0 = threadIdx.x * 2;                // fill rows for this thread
  v8f acc[4] = {};

  fill_vtile(vbuf, vb, r0, 0);             // prefill tile 0
  for (int kt = 0; kt < N_ / KT_; ++kt) {
    f16* cur = vbuf + (size_t)(kt & 1) * C_ * VLD_;
    wait_async0();                         // my fills for `cur` are done
    __syncthreads();                       // everyone's fills are done
    if (kt + 1 < N_ / KT_)                 // overlap next fill with compute
      fill_vtile(vbuf + (size_t)((kt + 1) & 1) * C_ * VLD_, vb, r0,
                 (kt + 1) * KT_);
    int kbase = kt * KT_;
#pragma unroll
    for (int kk = 0; kk < KT_; kk += 32) {
      v16h a = load_a16(S, SLD, 0, kbase + kk, lane);  // P from LDS
#pragma unroll
      for (int t = 0; t < 4; ++t) {
        const f16* p = cur + (size_t)(c0 + 16 * t + nl) * VLD_ + kk + 16 * h;
        v16h bm = join8(*(const v8h*)p, *(const v8h*)(p + 8));
        acc[t] = wmma16(a, bm, acc[t]);
      }
    }
    // no trailing barrier needed: `cur` is only refilled after the next
    // iteration's wait+barrier, which every wave reaches post-compute.
  }

  v8f scl;
#pragma unroll
  for (int vv = 0; vv < 8; ++vv) scl[vv] = stats[vv + 8 * h];
  float* ob = out + (size_t)b * C_ * N_ + i0 + 8 * h;
#pragma unroll
  for (int t = 0; t < 4; ++t) {
    int    c = c0 + 16 * t + nl;
    float* p = ob + (size_t)c * N_;
#pragma unroll
    for (int vv = 0; vv < 8; ++vv) p[vv] = acc[t][vv] * scl[vv];
  }
}

// ------------------------------- launcher ----------------------------------
extern "C" void kernel_launch(void* const* d_in, const int* in_sizes, int n_in,
                              void* d_out, int out_size, void* d_ws,
                              size_t ws_size, hipStream_t stream) {
  const float* x  = (const float*)d_in[0];
  const float* wq = (const float*)d_in[1];
  const float* bq = (const float*)d_in[2];
  const float* wk = (const float*)d_in[3];
  const float* bk = (const float*)d_in[4];
  const float* wv = (const float*)d_in[5];
  const float* bv = (const float*)d_in[6];
  float* out = (float*)d_out;

  char* ws = (char*)d_ws;
  size_t off = 0;
  f16* xT  = (f16*)(ws + off); off += (size_t)B_ * N_ * C_ * 2;
  f16* wqh = (f16*)(ws + off); off += (size_t)CQK_ * C_ * 2;
  f16* wkh = (f16*)(ws + off); off += (size_t)CQK_ * C_ * 2;
  f16* wvh = (f16*)(ws + off); off += (size_t)C_ * C_ * 2;
  f16* qh  = (f16*)(ws + off); off += (size_t)B_ * N_ * CQK_ * 2;
  f16* kh  = (f16*)(ws + off); off += (size_t)B_ * N_ * CQK_ * 2;
  f16* vh  = (f16*)(ws + off); off += (size_t)B_ * C_ * N_ * 2;

  cvt_f32_f16<<<(CQK_ * C_ + 255) / 256, 256, 0, stream>>>(wq, wqh, CQK_ * C_);
  cvt_f32_f16<<<(CQK_ * C_ + 255) / 256, 256, 0, stream>>>(wk, wkh, CQK_ * C_);
  cvt_f32_f16<<<(C_ * C_ + 255) / 256, 256, 0, stream>>>(wv, wvh, C_ * C_);

  transpose_cvt<<<dim3(N_ / 32, C_ / 32, B_), dim3(32, 8), 0, stream>>>(x, xT);

  proj_gemm<<<dim3(N_ / 128, 1, B_), 256, 0, stream>>>(xT, wqh, bq, qh, CQK_, 0);
  proj_gemm<<<dim3(N_ / 128, 1, B_), 256, 0, stream>>>(xT, wkh, bk, kh, CQK_, 0);
  proj_gemm<<<dim3(N_ / 128, C_ / 64, B_), 256, 0, stream>>>(xT, wvh, bv, vh, C_, 1);

  size_t smem = (size_t)16 * SLD * sizeof(f16) + 16 * sizeof(float)
              + (size_t)2 * C_ * VLD_ * sizeof(f16);
  (void)hipFuncSetAttribute((const void*)attention_fused,
                            hipFuncAttributeMaxDynamicSharedMemorySize,
                            (int)smem);
  attention_fused<<<dim3(N_ / 16, B_), 256, smem, stream>>>(qh, kh, vh, out);
}
